// YOWOF_9440338117111
// MI455X (gfx1250) — matrix operand order repaired and checked by reading, not compiled, not run on hardware
//
#include <hip/hip_runtime.h>
#include <hip/hip_bf16.h>
#include <math.h>

// ---------------- problem constants ----------------
#define HH        160
#define WW        160
#define PW        162            // padded width/height (+1 ring)
#define CIN       256
#define KTOT      2304           // 9 * 256
#define HWTOT     25600          // 160*160
#define NA        5              // anchors
#define NC        20             // classes
#define NALL      (HWTOT * NA)   // 128000
#define MCLS      112            // 100 cls rows padded to 7*16
#define MOR       32             // 5 obj + 20 reg rows padded to 2*16
#define TOPK_N    1000
#define SELCAP    4096
#define EXP_CLAMP_F    18.420680743952367f   // log(1e8)
#define SCALE_CLAMP_F  4.135166556742356f    // log(1000/16)
#define CONF_TH        0.05f
#define NMS_TH         0.6f

typedef __attribute__((ext_vector_type(16))) _Float16 v16h;
typedef __attribute__((ext_vector_type(8)))  float    v8f;
typedef unsigned long long u64;

union FragU { v16h h; uint4 q[2]; };

// CDNA5 async Global->LDS copy (ASYNCcnt-tracked), GV addressing mode.
__device__ __forceinline__ void async_b128(unsigned lds_byte_addr, const void* gaddr) {
    asm volatile("global_load_async_to_lds_b128 %0, %1, off"
                 :: "v"(lds_byte_addr), "v"(gaddr) : "memory");
}
__device__ __forceinline__ void wait_async0() {
    asm volatile("s_wait_asynccnt 0x0" ::: "memory");
}
// Generic LDS pointer: low 32 bits are the LDS byte offset on CDNA5.
__device__ __forceinline__ unsigned lds_addr(const void* p) {
    return (unsigned)(unsigned long long)p;
}

// ---------------- 1) pack features: f32 NCHW -> f16 padded channels-last ----
__global__ void yowof_pad_feats(const float* __restrict__ clsf,
                                const float* __restrict__ regf,
                                _Float16* __restrict__ outCls,
                                _Float16* __restrict__ outReg) {
    int t = blockIdx.x * blockDim.x + threadIdx.x;
    const int total = PW * PW * CIN;
    if (t >= total) return;
    int c     = t % CIN;
    int pixel = t / CIN;
    int px = pixel % PW, py = pixel / PW;
    float vc = 0.f, vr = 0.f;
    if (px >= 1 && px <= WW && py >= 1 && py <= HH) {
        int src = c * HWTOT + (py - 1) * WW + (px - 1);
        vc = clsf[src];
        vr = regf[src];
    }
    outCls[t] = (_Float16)vc;
    outReg[t] = (_Float16)vr;
}

// ---------------- 2) pack weights: OIHW f32 -> [row][tap*256+ci] f16 -------
__global__ void yowof_pack_weights(const float* __restrict__ cls_w,
                                   const float* __restrict__ obj_w,
                                   const float* __restrict__ reg_w,
                                   _Float16* __restrict__ Wcls,
                                   _Float16* __restrict__ Wor) {
    int t = blockIdx.x * blockDim.x + threadIdx.x;
    const int total = (MCLS + MOR) * KTOT;
    if (t >= total) return;
    int col = t % KTOT;
    int row = t / KTOT;
    int tap = col / CIN, ci = col % CIN;
    int srcOff = ci * 9 + tap;               // w[o][ci][ky][kx], tap = ky*3+kx
    if (row < MCLS) {
        float v = (row < NA * NC) ? cls_w[row * (CIN * 9) + srcOff] : 0.f;
        Wcls[row * KTOT + col] = (_Float16)v;
    } else {
        int r = row - MCLS;                  // 0..4 obj, 5..24 reg, 25..31 pad
        float v = 0.f;
        if (r < NA)           v = obj_w[r * (CIN * 9) + srcOff];
        else if (r < NA + 20) v = reg_w[(r - NA) * (CIN * 9) + srcOff];
        Wor[r * KTOT + col] = (_Float16)v;
    }
}

// ---------------- 3) implicit-GEMM 3x3 conv via WMMA -----------------------
// Block = 4 waves. Each wave owns TWO 16-column N-tiles of one output row
// (block covers 128 columns). Weights for the block's M-tile are staged in
// LDS per-ky slab (16x768 f16 = 24KB) with double-buffered async copies, so
// A fragments come from LDS (shared by all 4 waves / 8 accumulator tiles)
// while B fragments stream from global (L2-resident).
// A (16x32 f16): lane = M row (lane&15); K base = half*16, contiguous 16.
// B (32x16 f16): lane = N col (lane&15); K base = half*16, contiguous 16.
#define SLAB 768                 // 3 taps * 256 channels per ky row
__global__ void __launch_bounds__(128)
yowof_conv_gemm(const _Float16* __restrict__ Wmat,
                const _Float16* __restrict__ featPad,
                float* __restrict__ out) {
    __shared__ _Float16 As[2][16 * SLAB];    // 2 x 24KB double buffer

    const int tid  = threadIdx.x;
    const int lane = tid & 31;
    const int wave = tid >> 5;
    const int half = lane >> 4;     // 0 or 1 (lane group)
    const int l16  = lane & 15;

    const int pair  = blockIdx.x * 4 + wave;   // 0..799, 2 N-tiles each
    const int y     = pair / 5;                // 5 pairs (160 cols) per row
    const int x0    = (pair % 5) * 32;
    const int mbase = blockIdx.y * 16;

    // stage ky=0 weight slab: 16 rows x 768 halves = 1536 b128 chunks
#pragma unroll
    for (int it = 0; it < 12; ++it) {
        int i = tid + it * 128;
        int r = i / 96, c = i % 96;
        const void* g = Wmat + (size_t)(mbase + r) * KTOT + c * 8;
        async_b128(lds_addr(&As[0][r * SLAB + c * 8]), g);
    }

    v8f acc0 = {}, acc1 = {};
    for (int ky = 0; ky < 3; ++ky) {
        wait_async0();
        __syncthreads();
        if (ky < 2) {   // prefetch next slab into the other buffer
#pragma unroll
            for (int it = 0; it < 12; ++it) {
                int i = tid + it * 128;
                int r = i / 96, c = i % 96;
                const void* g = Wmat + (size_t)(mbase + r) * KTOT
                                     + (ky + 1) * SLAB + c * 8;
                async_b128(lds_addr(&As[(ky + 1) & 1][r * SLAB + c * 8]), g);
            }
        }
        const _Float16* Acur = As[ky & 1];
#pragma unroll
        for (int kx = 0; kx < 3; ++kx) {
            const _Float16* bpix =
                featPad + (size_t)((y + ky) * PW + (x0 + l16 + kx)) * CIN + half * 16;
#pragma unroll
            for (int ch = 0; ch < 8; ++ch) {   // K chunks of 32 channels
                const int ab = l16 * SLAB + kx * CIN + ch * 32 + half * 16;
                FragU a, b0, b1;
                a.q[0]  = *(const uint4*)(Acur + ab);         // ds_load_b128
                a.q[1]  = *(const uint4*)(Acur + ab + 8);
                b0.q[0] = *(const uint4*)(bpix + ch * 32);
                b0.q[1] = *(const uint4*)(bpix + ch * 32 + 8);
                b1.q[0] = *(const uint4*)(bpix + 16 * CIN + ch * 32);
                b1.q[1] = *(const uint4*)(bpix + 16 * CIN + ch * 32 + 8);
                acc0 = __builtin_amdgcn_wmma_f32_16x16x32_f16(
                           false, a.h, false, b0.h, (short)0, acc0, false, false);
                acc1 = __builtin_amdgcn_wmma_f32_16x16x32_f16(
                           false, a.h, false, b1.h, (short)0, acc1, false, false);
            }
        }
        __syncthreads();   // all waves done reading before slab is refilled
    }
    // D layout: VGPR j -> M = j + half*8, N = lane&15
    const int p0 = y * WW + x0 + l16;
#pragma unroll
    for (int j = 0; j < 8; ++j) {
        out[(size_t)(mbase + j + half * 8) * HWTOT + p0]      = acc0[j];
        out[(size_t)(mbase + j + half * 8) * HWTOT + p0 + 16] = acc1[j];
    }
}

// ---------------- 4) per-(pixel,anchor) score normalize + argmax -----------
__global__ void yowof_head_post(const float* __restrict__ convCls,
                                const float* __restrict__ convOR,
                                const float* __restrict__ obj_b,
                                const float* __restrict__ cls_b,
                                const float* __restrict__ reg_b,
                                float* __restrict__ scoresAll,
                                int*   __restrict__ labelsAll,
                                float* __restrict__ regAll) {
    int t = blockIdx.x * blockDim.x + threadIdx.x;
    if (t >= NALL) return;
    int a = t % NA, p = t / NA;
    float o  = convOR[(size_t)a * HWTOT + p] + obj_b[a];
    float eo = expf(fminf(o, EXP_CLAMP_F));
    float best = -1.f; int bl = 0;
#pragma unroll 4
    for (int c = 0; c < NC; ++c) {
        int ch = a * NC + c;
        float cl   = convCls[(size_t)ch * HWTOT + p] + cls_b[ch];
        float norm = cl + o - logf(1.f + expf(fminf(cl, EXP_CLAMP_F)) + eo);
        float prob = 1.f / (1.f + expf(-norm));
        if (prob > best) { best = prob; bl = c; }
    }
    scoresAll[t] = best;
    labelsAll[t] = bl;
#pragma unroll
    for (int k = 0; k < 4; ++k)
        regAll[(size_t)t * 4 + k] =
            convOR[(size_t)(NA + a * 4 + k) * HWTOT + p] + reg_b[a * 4 + k];
}

// ---------------- 5) top-k selection: radix threshold + compact + sort -----
__global__ void yowof_init_select(unsigned* __restrict__ hist,
                                  unsigned* __restrict__ meta,
                                  u64* __restrict__ sel) {
    int t = blockIdx.x * blockDim.x + threadIdx.x;
    if (t < SELCAP) { hist[t] = 0u; sel[t] = 0ull; }
    if (t < 4) meta[t] = 0u;
}

__global__ void yowof_histo(const float* __restrict__ scoresAll,
                            unsigned* __restrict__ hist) {
    int t = blockIdx.x * blockDim.x + threadIdx.x;
    if (t >= NALL) return;
    atomicAdd(&hist[__float_as_uint(scoresAll[t]) >> 20], 1u);
}

__global__ void yowof_threshold(const unsigned* __restrict__ hist,
                                unsigned* __restrict__ meta) {
    if (blockIdx.x != 0 || threadIdx.x != 0) return;
    unsigned cum = 0, thr = 0;
    for (int b = SELCAP - 1; b >= 0; --b) {
        cum += hist[b];
        if (cum >= (unsigned)TOPK_N) { thr = (unsigned)b << 20; break; }
    }
    meta[1] = thr;
}

__global__ void yowof_compact(const float* __restrict__ scoresAll,
                              unsigned* __restrict__ meta,
                              u64* __restrict__ sel) {
    int t = blockIdx.x * blockDim.x + threadIdx.x;
    if (t >= NALL) return;
    unsigned key = __float_as_uint(scoresAll[t]);
    if (key >= meta[1]) {
        unsigned pos = atomicAdd(&meta[0], 1u);
        if (pos < SELCAP)
            sel[pos] = ((u64)key << 32) | (unsigned)(~(unsigned)t);  // desc score, asc idx
    }
}

__global__ void __launch_bounds__(1024) yowof_bitonic(u64* __restrict__ sel) {
    __shared__ u64 s[SELCAP];
    for (int i = threadIdx.x; i < SELCAP; i += 1024) s[i] = sel[i];
    __syncthreads();
    for (int k = 2; k <= SELCAP; k <<= 1) {
        for (int j = k >> 1; j > 0; j >>= 1) {
            for (int t = threadIdx.x; t < SELCAP; t += 1024) {
                int ixj = t ^ j;
                if (ixj > t) {
                    u64 A = s[t], B = s[ixj];
                    bool desc = ((t & k) == 0);
                    if (desc ? (A < B) : (A > B)) { s[t] = B; s[ixj] = A; }
                }
            }
            __syncthreads();
        }
    }
    for (int i = threadIdx.x; i < SELCAP; i += 1024) sel[i] = s[i];
}

// ---------------- 6) decode top-1000 boxes ---------------------------------
__global__ void yowof_decode(const u64* __restrict__ sel,
                             const int* __restrict__ labelsAll,
                             const float* __restrict__ regAll,
                             float* __restrict__ tScore,
                             int*   __restrict__ tLabel,
                             float* __restrict__ tBox,
                             float* __restrict__ tBoxOff,
                             unsigned* __restrict__ validArr) {
    int i = blockIdx.x * blockDim.x + threadIdx.x;
    if (i >= TOPK_N) return;
    const float ANCH[NA] = {32.f, 64.f, 128.f, 256.f, 512.f};
    u64 C = sel[i];
    float score = __uint_as_float((unsigned)(C >> 32));
    int idx = (int)(~(unsigned)(C & 0xFFFFFFFFull));
    if (idx < 0 || idx >= NALL) idx = 0;
    int a = idx % NA, p = idx / NA;
    int x = p % WW,  y = p / WW;
    float ax = (x + 0.5f) * 8.f, ay = (y + 0.5f) * 8.f;
    float aw = ANCH[a], ah = ANCH[a];
    float r0 = regAll[(size_t)idx * 4 + 0];
    float r1 = regAll[(size_t)idx * 4 + 1];
    float r2 = regAll[(size_t)idx * 4 + 2];
    float r3 = regAll[(size_t)idx * 4 + 3];
    float cx = ax + fminf(fmaxf(r0 * aw, -32.f), 32.f);
    float cy = ay + fminf(fmaxf(r1 * ah, -32.f), 32.f);
    float w  = aw * expf(fminf(r2, SCALE_CLAMP_F));
    float h  = ah * expf(fminf(r3, SCALE_CLAMP_F));
    float bx1 = fminf(fmaxf((cx - 0.5f * w) * (1.f / 1280.f), 0.f), 1.f);
    float by1 = fminf(fmaxf((cy - 0.5f * h) * (1.f / 1280.f), 0.f), 1.f);
    float bx2 = fminf(fmaxf((cx + 0.5f * w) * (1.f / 1280.f), 0.f), 1.f);
    float by2 = fminf(fmaxf((cy + 0.5f * h) * (1.f / 1280.f), 0.f), 1.f);
    int lab = labelsAll[idx];
    float lo = (float)lab * 2.f;
    tScore[i] = score;
    tLabel[i] = lab;
    tBox[i * 4 + 0] = bx1; tBox[i * 4 + 1] = by1;
    tBox[i * 4 + 2] = bx2; tBox[i * 4 + 3] = by2;
    tBoxOff[i * 4 + 0] = bx1 + lo; tBoxOff[i * 4 + 1] = by1 + lo;
    tBoxOff[i * 4 + 2] = bx2 + lo; tBoxOff[i * 4 + 3] = by2 + lo;
    validArr[i] = (score >= CONF_TH) ? 1u : 0u;
}

// ---------------- 7) pairwise IoU suppression bitmask ----------------------
__global__ void yowof_iou_mask(const float* __restrict__ bo,
                               unsigned* __restrict__ mask) {
    int t = blockIdx.x * blockDim.x + threadIdx.x;   // t = i*32 + word
    if (t >= TOPK_N * 32) return;
    int w = t % 32, i = t / 32;
    float x1 = bo[i * 4], y1 = bo[i * 4 + 1], x2 = bo[i * 4 + 2], y2 = bo[i * 4 + 3];
    float ai = (x2 - x1) * (y2 - y1);
    unsigned m = 0;
    for (int b = 0; b < 32; ++b) {
        int j = w * 32 + b;
        if (j >= TOPK_N || j <= i) continue;
        float jx1 = bo[j * 4], jy1 = bo[j * 4 + 1], jx2 = bo[j * 4 + 2], jy2 = bo[j * 4 + 3];
        float xx1 = fmaxf(x1, jx1), yy1 = fmaxf(y1, jy1);
        float xx2 = fminf(x2, jx2), yy2 = fminf(y2, jy2);
        float inter = fmaxf(1e-28f, xx2 - xx1) * fmaxf(1e-28f, yy2 - yy1);
        float aj = (jx2 - jx1) * (jy2 - jy1);
        float ovr = inter / (ai + aj - inter + 1e-14f);
        if (ovr > NMS_TH) m |= 1u << b;
    }
    mask[t] = m;
}

// ---------------- 8) greedy NMS scan, one wave32, bit-words in LDS ---------
__global__ void yowof_nms_scan(const unsigned* __restrict__ mask,
                               const unsigned* __restrict__ validIn,
                               unsigned* __restrict__ keepWords) {
    __shared__ unsigned vw[32];
    int lane = threadIdx.x;          // 32 threads
    unsigned w = 0;
    for (int b = 0; b < 32; ++b) {
        int j = lane * 32 + b;
        if (j < TOPK_N && validIn[j]) w |= 1u << b;
    }
    vw[lane] = w;
    __syncthreads();
    for (int i = 0; i < TOPK_N; ++i) {
        unsigned bit = (vw[i >> 5] >> (i & 31)) & 1u;
        if (bit) vw[lane] &= ~mask[i * 32 + lane];   // mask only has j > i
        __syncthreads();
    }
    keepWords[lane] = vw[lane];
}

// ---------------- 9) finalize outputs --------------------------------------
__global__ void yowof_finalize(const float* __restrict__ tScore,
                               const int*   __restrict__ tLabel,
                               const float* __restrict__ tBox,
                               const unsigned* __restrict__ keepWords,
                               float* __restrict__ out) {
    int i = blockIdx.x * blockDim.x + threadIdx.x;
    if (i >= TOPK_N) return;
    float k = ((keepWords[i >> 5] >> (i & 31)) & 1u) ? 1.f : 0.f;
    out[i * 5 + 0] = tScore[i] * k;
    out[i * 5 + 1] = tBox[i * 4 + 0] * k;
    out[i * 5 + 2] = tBox[i * 4 + 1] * k;
    out[i * 5 + 3] = tBox[i * 4 + 2] * k;
    out[i * 5 + 4] = tBox[i * 4 + 3] * k;
    out[5 * TOPK_N + i] = (float)tLabel[i] * k;
    out[6 * TOPK_N + i] = k;
}

// ---------------- host-side orchestration ----------------------------------
extern "C" void kernel_launch(void* const* d_in, const int* in_sizes, int n_in,
                              void* d_out, int out_size, void* d_ws, size_t ws_size,
                              hipStream_t stream) {
    const float* cls_feats = (const float*)d_in[0];
    const float* reg_feats = (const float*)d_in[1];
    const float* obj_w     = (const float*)d_in[2];
    const float* obj_b     = (const float*)d_in[3];
    const float* cls_w     = (const float*)d_in[4];
    const float* cls_b     = (const float*)d_in[5];
    const float* reg_w     = (const float*)d_in[6];
    const float* reg_b     = (const float*)d_in[7];
    float* out = (float*)d_out;

    char* ws = (char*)d_ws;
    size_t off = 0;
    auto carve = [&](size_t bytes) -> char* {
        char* p = ws + off;
        off = (off + bytes + 255) & ~(size_t)255;
        return p;
    };
    _Float16* featCls = (_Float16*)carve((size_t)PW * PW * CIN * 2);
    _Float16* featReg = (_Float16*)carve((size_t)PW * PW * CIN * 2);
    _Float16* Wcls    = (_Float16*)carve((size_t)MCLS * KTOT * 2);
    _Float16* Wor     = (_Float16*)carve((size_t)MOR * KTOT * 2);
    float*    convCls = (float*)carve((size_t)MCLS * HWTOT * 4);
    float*    convOR  = (float*)carve((size_t)MOR * HWTOT * 4);
    float*    scoresA = (float*)carve((size_t)NALL * 4);
    int*      labelsA = (int*)carve((size_t)NALL * 4);
    float*    regA    = (float*)carve((size_t)NALL * 4 * 4);
    unsigned* hist    = (unsigned*)carve(SELCAP * 4);
    unsigned* meta    = (unsigned*)carve(64);
    u64*      sel     = (u64*)carve(SELCAP * 8);
    float*    tScore  = (float*)carve(TOPK_N * 4);
    int*      tLabel  = (int*)carve(TOPK_N * 4);
    float*    tBox    = (float*)carve(TOPK_N * 4 * 4);
    float*    tBoxOff = (float*)carve(TOPK_N * 4 * 4);
    unsigned* validA  = (unsigned*)carve(TOPK_N * 4);
    unsigned* supMask = (unsigned*)carve((size_t)TOPK_N * 32 * 4);
    unsigned* keepW   = (unsigned*)carve(32 * 4);

    // 1) pack features (padded, channels-last, f16)
    {
        int total = PW * PW * CIN;
        yowof_pad_feats<<<(total + 255) / 256, 256, 0, stream>>>(
            cls_feats, reg_feats, featCls, featReg);
    }
    // 2) pack weights
    {
        int total = (MCLS + MOR) * KTOT;
        yowof_pack_weights<<<(total + 255) / 256, 256, 0, stream>>>(
            cls_w, obj_w, reg_w, Wcls, Wor);
    }
    // 3) the two WMMA implicit-GEMM convolutions (LDS-staged weights)
    yowof_conv_gemm<<<dim3(200, MCLS / 16), 128, 0, stream>>>(Wcls, featCls, convCls);
    yowof_conv_gemm<<<dim3(200, MOR  / 16), 128, 0, stream>>>(Wor,  featReg, convOR);
    // 4) score normalization + argmax
    yowof_head_post<<<(NALL + 255) / 256, 256, 0, stream>>>(
        convCls, convOR, obj_b, cls_b, reg_b, scoresA, labelsA, regA);
    // 5) top-k
    yowof_init_select<<<(SELCAP + 255) / 256, 256, 0, stream>>>(hist, meta, sel);
    yowof_histo<<<(NALL + 255) / 256, 256, 0, stream>>>(scoresA, hist);
    yowof_threshold<<<1, 32, 0, stream>>>(hist, meta);
    yowof_compact<<<(NALL + 255) / 256, 256, 0, stream>>>(scoresA, meta, sel);
    yowof_bitonic<<<1, 1024, 0, stream>>>(sel);
    // 6) decode
    yowof_decode<<<(TOPK_N + 255) / 256, 256, 0, stream>>>(
        sel, labelsA, regA, tScore, tLabel, tBox, tBoxOff, validA);
    // 7) IoU mask + 8) greedy NMS
    yowof_iou_mask<<<(TOPK_N * 32 + 255) / 256, 256, 0, stream>>>(tBoxOff, supMask);
    yowof_nms_scan<<<1, 32, 0, stream>>>(supMask, validA, keepW);
    // 9) outputs
    yowof_finalize<<<(TOPK_N + 255) / 256, 256, 0, stream>>>(
        tScore, tLabel, tBox, keepW, out);
}
